// operator_iso_18081812316514
// MI455X (gfx1250) — matrix operand order repaired and checked by reading, not compiled, not run on hardware
//
#include <hip/hip_runtime.h>
#include <hip/hip_bf16.h>

// ---------------------------------------------------------------------------
// Shapes
// ---------------------------------------------------------------------------
#define BATCH    128
#define SPATIAL  1024
#define CHANNELS 128
#define OP_DIM   256

// d_out layout (floats): tau | Phi | Lambda | M
#define TAU_N   (BATCH * OP_DIM * OP_DIM)     // 8388608
#define PHI_N   (SPATIAL * OP_DIM)            // 262144
#define TAU_OFF 0
#define PHI_OFF (TAU_N)
#define LAM_OFF (TAU_N + PHI_N)               // 8650752
#define M_OFF   (LAM_OFF + OP_DIM)            // 8651008

// GEMM tiling: block tile 128x64x32, 8 waves, each wave owns 32x32 (4 WMMAs)
#define BM 128
#define BN 64
#define BK 32
#define LDSS 40   // padded LDS row stride in ushorts; 40*2=80B keeps b128 alignment

typedef __attribute__((ext_vector_type(16))) __bf16 v16bf;
typedef __attribute__((ext_vector_type(8)))  float  v8f;
typedef unsigned short ushort_t;

union FragAB { v16bf v; unsigned u[8]; };

__device__ __forceinline__ float bf2f(ushort_t h) {
    return __uint_as_float(((unsigned)h) << 16);
}
__device__ __forceinline__ ushort_t f2bf(float f) {
    unsigned u = __float_as_uint(f);
    u += 0x7FFFu + ((u >> 16) & 1u);          // round-to-nearest-even
    return (ushort_t)(u >> 16);
}

// Epilogue selector
//  0 : store bf16
//  1 : v = 1.5*Extra - 0.5*v  -> bf16            (Newton-Schulz step)
//  2 : v = 1.5*Extra - 0.5*v  -> f32             (final NS step into d_out)
//  3 : v *= exp(-|Lam[m]-Lam[n]|) -> bf16, + per-batch atomic sum(v^2)
//
// TA : A(m,k) = Asrc[k*lda + m]   (else Asrc[m*lda + k])
// TB : B(k,n) = Bsrc[n*ldb + k]   (else Bsrc[k*ldb + n])
// BF32B : B source is f32 (converted to bf16 while staging)
// All M,N,K used here are multiples of 128/64/32 -> no bounds checks.
template<int TA, int TB, int BF32B, int EPI>
__global__ __launch_bounds__(256) void wmma_gemm(
    const ushort_t* __restrict__ A, long long sA, int lda,
    const void* __restrict__ B, long long sB, int ldb,
    int K,
    ushort_t* __restrict__ Cbf, long long sC,
    float* __restrict__ Cf,
    const ushort_t* __restrict__ Extra,
    int ldc,
    const float* __restrict__ Lam, float* __restrict__ sumsq)
{
    __shared__ __align__(16) ushort_t As[2][BM * LDSS];
    __shared__ __align__(16) ushort_t Bs[2][BN * LDSS];   // n-major: Bs[n][k]

    const int tid  = threadIdx.x;
    const int lane = tid & 31;
    const int w    = tid >> 5;
    const int wm   = (w >> 1) * 32;    // wave M offset in block tile
    const int wn   = (w & 1) * 32;     // wave N offset in block tile
    const long long batch = blockIdx.z;
    const int mBase = blockIdx.y * BM;
    const int nBase = blockIdx.x * BN;

    const ushort_t* Ab = A + batch * sA;
    const float*    Bf = (const float*)B + batch * sB;
    const ushort_t* Bh = (const ushort_t*)B + batch * sB;

    // ---- staging helpers (all index math is compile-time specialized) ----
    uint4  ra[2];        // A: 2 chunks of 8 bf16
    uint4  rb;           // B bf16: 1 chunk of 8 bf16
    float4 rbf[2];       // B f32 : 2 chunks of 4 f32

    auto fetch = [&](int k0) {
        #pragma unroll
        for (int i = 0; i < 2; ++i) {
            const int c = tid + i * 256;               // 512 chunks of 8
            long long ga;
            if constexpr (!TA)
                ga = (long long)(mBase + (c >> 2)) * lda + (k0 + (c & 3) * 8);
            else
                ga = (long long)(k0 + (c >> 4)) * lda + (mBase + (c & 15) * 8);
            ra[i] = *(const uint4*)(Ab + ga);
        }
        if constexpr (BF32B) {
            #pragma unroll
            for (int i = 0; i < 2; ++i) {
                const int c = tid + i * 256;           // 512 chunks of 4
                rbf[i] = *(const float4*)(Bf + (long long)(k0 + (c >> 4)) * ldb
                                             + (nBase + (c & 15) * 4));
            }
        } else if constexpr (!TB) {
            rb = *(const uint4*)(Bh + (long long)(k0 + (tid >> 3)) * ldb
                                    + (nBase + (tid & 7) * 8));
        } else {
            rb = *(const uint4*)(Bh + (long long)(nBase + (tid >> 2)) * ldb
                                    + (k0 + (tid & 3) * 8));
        }
    };

    auto stage = [&](int bi) {
        ushort_t* Asb = As[bi];
        ushort_t* Bsb = Bs[bi];
        #pragma unroll
        for (int i = 0; i < 2; ++i) {
            const int c = tid + i * 256;
            if constexpr (!TA) {
                *(uint4*)&Asb[(c >> 2) * LDSS + (c & 3) * 8] = ra[i];
            } else {
                const ushort_t* p = (const ushort_t*)&ra[i];
                const int k = c >> 4, seg = c & 15;
                #pragma unroll
                for (int j = 0; j < 8; ++j)
                    Asb[(seg * 8 + j) * LDSS + k] = p[j];
            }
        }
        if constexpr (BF32B) {
            #pragma unroll
            for (int i = 0; i < 2; ++i) {
                const int c = tid + i * 256;
                const int k = c >> 4, seg = c & 15;
                const float* q = (const float*)&rbf[i];
                #pragma unroll
                for (int j = 0; j < 4; ++j)
                    Bsb[(seg * 4 + j) * LDSS + k] = f2bf(q[j]);
            }
        } else if constexpr (!TB) {
            const ushort_t* p = (const ushort_t*)&rb;
            const int k = tid >> 3, seg = tid & 7;
            #pragma unroll
            for (int j = 0; j < 8; ++j)
                Bsb[(seg * 8 + j) * LDSS + k] = p[j];
        } else {
            *(uint4*)&Bsb[(tid >> 2) * LDSS + (tid & 3) * 8] = rb;
        }
    };

    // ---- main pipeline: double-buffered LDS, fetch(t+1) overlaps WMMA(t) ----
    v8f cacc[4] = {};     // (m0,n0) (m0,n0+16) (m0+16,n0) (m0+16,n0+16)
    const int nK = K / BK;

    fetch(0);
    stage(0);

    for (int kt = 0; kt < nK; ++kt) {
        __syncthreads();                       // buffer kt&1 ready for all waves
        if (kt + 1 < nK) fetch((kt + 1) * BK); // issue global loads early

        // fragment packing per ISA 7.12.2 (wave32, bf16 A 16x32 / B 32x16)
        const int bi = kt & 1;
        FragAB fa0, fa1, fb0, fb1;
        {
            const int m  = lane & 15;
            const int kh = (lane >> 4) * 8;
            const ushort_t* a0 = &As[bi][(wm + m) * LDSS];
            const ushort_t* a1 = a0 + 16 * LDSS;
            #pragma unroll
            for (int p = 0; p < 4; ++p) {
                fa0.u[p]     = *(const unsigned*)(a0 + kh + 2 * p);
                fa0.u[p + 4] = *(const unsigned*)(a0 + 16 + kh + 2 * p);
                fa1.u[p]     = *(const unsigned*)(a1 + kh + 2 * p);
                fa1.u[p + 4] = *(const unsigned*)(a1 + 16 + kh + 2 * p);
            }
            const int kb = (lane >> 4) * 8;
            const ushort_t* b0 = &Bs[bi][(wn + (lane & 15)) * LDSS + kb];
            const ushort_t* b1 = b0 + 16 * LDSS;
            #pragma unroll
            for (int p = 0; p < 4; ++p) {
                fb0.u[p]     = *(const unsigned*)(b0 + 2 * p);
                fb0.u[p + 4] = *(const unsigned*)(b0 + 16 + 2 * p);
                fb1.u[p]     = *(const unsigned*)(b1 + 2 * p);
                fb1.u[p + 4] = *(const unsigned*)(b1 + 16 + 2 * p);
            }
        }
        cacc[0] = __builtin_amdgcn_wmma_f32_16x16x32_bf16(false, fa0.v, false, fb0.v,
                                                          (short)0, cacc[0], false, false);
        cacc[1] = __builtin_amdgcn_wmma_f32_16x16x32_bf16(false, fa0.v, false, fb1.v,
                                                          (short)0, cacc[1], false, false);
        cacc[2] = __builtin_amdgcn_wmma_f32_16x16x32_bf16(false, fa1.v, false, fb0.v,
                                                          (short)0, cacc[2], false, false);
        cacc[3] = __builtin_amdgcn_wmma_f32_16x16x32_bf16(false, fa1.v, false, fb1.v,
                                                          (short)0, cacc[3], false, false);

        __syncthreads();                       // everyone done reading buf bi^1
        if (kt + 1 < nK) stage(bi ^ 1);        // waits loads, fills next buffer
    }

    // ---- epilogue; C layout: elem r -> m = r + 8*(lane>=16), n = lane&15 ----
    float localss = 0.f;
    #pragma unroll
    for (int ti = 0; ti < 4; ++ti) {
        v8f cc = cacc[ti];
        const int mt = wm + (ti >> 1) * 16;
        const int nt = wn + (ti & 1) * 16;
        #pragma unroll
        for (int r = 0; r < 8; ++r) {
            const int m = mBase + mt + r + ((lane >> 4) << 3);
            const int n = nBase + nt + (lane & 15);
            float v = cc[r];
            if constexpr (EPI == 3) v *= __expf(-fabsf(Lam[m] - Lam[n]));
            const long long cidx = batch * sC + (long long)m * ldc + n;
            if constexpr (EPI == 1 || EPI == 2)
                v = 1.5f * bf2f(Extra[cidx]) - 0.5f * v;
            if constexpr (EPI == 2) Cf[cidx] = v;
            else                    Cbf[cidx] = f2bf(v);
            if constexpr (EPI == 3) localss += v * v;
        }
    }
    if constexpr (EPI == 3) atomicAdd(&sumsq[blockIdx.z], localss);
}

// ---------------------------------------------------------------------------
// Small kernels
// ---------------------------------------------------------------------------

// M, sqrt(M), 1/sqrt(M); Lambda = cumsum(Lambda_param^2) (argsort is identity
// since a cumsum of squares is nondecreasing); zero the reduction scratch.
__global__ __launch_bounds__(1024) void prep_kernel(
    const float* __restrict__ Mp, const float* __restrict__ Lp,
    float* __restrict__ outM, float* __restrict__ outLam,
    float* __restrict__ MR, float* __restrict__ MRinv,
    float* __restrict__ sum0, float* __restrict__ sumB)
{
    const int t = threadIdx.x;
    __shared__ float s[OP_DIM];
    {
        float m = Mp[t] * Mp[t] + 1e-8f;
        outM[t] = m;
        float r = sqrtf(m);
        MR[t] = r;
        MRinv[t] = 1.0f / r;
    }
    if (t < OP_DIM) s[t] = Lp[t] * Lp[t];
    __syncthreads();
    for (int off = 1; off < OP_DIM; off <<= 1) {
        float v = 0.f;
        if (t < OP_DIM && t >= off) v = s[t - off];
        __syncthreads();
        if (t < OP_DIM) s[t] += v;
        __syncthreads();
    }
    if (t < OP_DIM) outLam[t] = s[t];
    if (t == 0) *sum0 = 0.f;
    if (t < BATCH) sumB[t] = 0.f;
}

// sum of squares of X0 = sqrt(M)[s] * Phi_param  (Frobenius for NS scaling)
__global__ __launch_bounds__(256) void x0_sumsq(
    const float* __restrict__ Phip, const float* __restrict__ MR,
    float* __restrict__ sum0)
{
    const int i = blockIdx.x * 256 + threadIdx.x;   // < SPATIAL*OP_DIM
    const int srow = i >> 8;
    float v = MR[srow] * Phip[i];
    __shared__ float red[256];
    red[threadIdx.x] = v * v;
    __syncthreads();
    for (int o = 128; o > 0; o >>= 1) {
        if (threadIdx.x < o) red[threadIdx.x] += red[threadIdx.x + o];
        __syncthreads();
    }
    if (threadIdx.x == 0) atomicAdd(sum0, red[0]);
}

// Y0 = X0 / ||X0||_F  -> bf16 (guarantees ||Y0||_2 <= 1 for Newton-Schulz)
__global__ __launch_bounds__(256) void x0_scale(
    const float* __restrict__ Phip, const float* __restrict__ MR,
    const float* __restrict__ sum0, ushort_t* __restrict__ Ybf)
{
    const int i = blockIdx.x * 256 + threadIdx.x;
    const int srow = i >> 8;
    float inv = rsqrtf(*sum0 + 1e-30f);
    Ybf[i] = f2bf(MR[srow] * Phip[i] * inv);
}

// Phi = (1/sqrt(M)) * Y  (f32 output)   and   Psi = M * Phi  (bf16, so that
// Phi^T (M*X) == Psi^T X in the batched GEMMs)
__global__ __launch_bounds__(256) void phi_psi(
    const ushort_t* __restrict__ Ybf, const float* __restrict__ MRinv,
    const float* __restrict__ Mfull, float* __restrict__ outPhi,
    ushort_t* __restrict__ Psibf)
{
    const int i = blockIdx.x * 256 + threadIdx.x;
    const int srow = i >> 8;
    float phi = MRinv[srow] * bf2f(Ybf[i]);
    outPhi[i] = phi;
    Psibf[i] = f2bf(Mfull[srow] * phi);
}

// Z0[b] = corr[b] / ||corr[b]||_F  -> bf16
__global__ __launch_bounds__(256) void z0_scale(
    const ushort_t* __restrict__ corrbf, const float* __restrict__ sumB,
    ushort_t* __restrict__ Zbf)
{
    const long long i = (long long)blockIdx.x * 256 + threadIdx.x; // < TAU_N
    const int b = (int)(i >> 16);
    Zbf[i] = f2bf(bf2f(corrbf[i]) * rsqrtf(sumB[b] + 1e-30f));
}

// ---------------------------------------------------------------------------
// Host launcher
// ---------------------------------------------------------------------------
extern "C" void kernel_launch(void* const* d_in, const int* in_sizes, int n_in,
                              void* d_out, int out_size, void* d_ws, size_t ws_size,
                              hipStream_t stream)
{
    const float* dA   = (const float*)d_in[0];
    const float* dB   = (const float*)d_in[1];
    const float* dMp  = (const float*)d_in[2];
    const float* dPhp = (const float*)d_in[3];
    const float* dLp  = (const float*)d_in[4];

    float* out    = (float*)d_out;
    float* outTau = out + TAU_OFF;
    float* outPhi = out + PHI_OFF;
    float* outLam = out + LAM_OFF;
    float* outM   = out + M_OFF;

    // workspace carve (256B aligned)
    char* wsp = (char*)d_ws;
    auto alloc = [&](size_t bytes) -> char* {
        char* p = wsp;
        wsp += (bytes + 255) & ~(size_t)255;
        return p;
    };
    float* MR    = (float*)alloc(SPATIAL * 4);
    float* MRinv = (float*)alloc(SPATIAL * 4);
    float* sum0  = (float*)alloc(4);
    float* sumB  = (float*)alloc(BATCH * 4);
    ushort_t* Ybf   = (ushort_t*)alloc((size_t)SPATIAL * OP_DIM * 2);
    ushort_t* Ybf2  = (ushort_t*)alloc((size_t)SPATIAL * OP_DIM * 2);
    ushort_t* Gbf   = (ushort_t*)alloc((size_t)OP_DIM * OP_DIM * 2);
    ushort_t* Psibf = (ushort_t*)alloc((size_t)SPATIAL * OP_DIM * 2);
    ushort_t* PTAbf = (ushort_t*)alloc((size_t)BATCH * OP_DIM * CHANNELS * 2);
    ushort_t* PTBbf = (ushort_t*)alloc((size_t)BATCH * OP_DIM * CHANNELS * 2);
    ushort_t* corrbf= (ushort_t*)alloc((size_t)TAU_N * 2);
    ushort_t* Zbf   = (ushort_t*)alloc((size_t)TAU_N * 2);
    ushort_t* Zbf2  = (ushort_t*)alloc((size_t)TAU_N * 2);
    ushort_t* Gbbf  = (ushort_t*)alloc((size_t)TAU_N * 2);

    const long long sPT  = (long long)OP_DIM * CHANNELS;  // 32768
    const long long sSq  = (long long)OP_DIM * OP_DIM;    // 65536

    // 1) scalars: M, sqrt/rsqrt, Lambda cumsum, zero reductions
    prep_kernel<<<1, 1024, 0, stream>>>(dMp, dLp, outM, outLam, MR, MRinv, sum0, sumB);

    // 2) Frobenius-normalized start for Phi Newton-Schulz
    x0_sumsq<<<(SPATIAL * OP_DIM) / 256, 256, 0, stream>>>(dPhp, MR, sum0);
    x0_scale<<<(SPATIAL * OP_DIM) / 256, 256, 0, stream>>>(dPhp, MR, sum0, Ybf);

    // 3) Newton-Schulz polar of 1024x256: Y <- 1.5Y - 0.5*Y*(Y^T Y)
    ushort_t* cur = Ybf;
    ushort_t* nxt = Ybf2;
    for (int it = 0; it < 10; ++it) {
        // G = Y^T Y  (256x256, K=1024)
        wmma_gemm<1,0,0,0><<<dim3(OP_DIM / BN, OP_DIM / BM, 1), 256, 0, stream>>>(
            cur, 0, OP_DIM, cur, 0, OP_DIM, SPATIAL,
            Gbf, 0, nullptr, nullptr, OP_DIM, nullptr, nullptr);
        // Ynext = 1.5Y - 0.5*(Y*G)  (1024x256, K=256)
        wmma_gemm<0,0,0,1><<<dim3(OP_DIM / BN, SPATIAL / BM, 1), 256, 0, stream>>>(
            cur, 0, OP_DIM, Gbf, 0, OP_DIM, OP_DIM,
            nxt, 0, nullptr, cur, OP_DIM, nullptr, nullptr);
        ushort_t* t = cur; cur = nxt; nxt = t;
    }

    // 4) Phi (f32 out) and Psi = M*Phi (bf16)
    phi_psi<<<(SPATIAL * OP_DIM) / 256, 256, 0, stream>>>(cur, MRinv, outM, outPhi, Psibf);

    // 5) PhiTMB / PhiTMA:  Psi^T (256x1024) @ X[b] (1024x128), f32 B staged->bf16
    wmma_gemm<1,0,1,0><<<dim3(CHANNELS / BN, OP_DIM / BM, BATCH), 256, 0, stream>>>(
        Psibf, 0, OP_DIM,
        (const void*)dB, (long long)SPATIAL * CHANNELS, CHANNELS, SPATIAL,
        PTBbf, sPT, nullptr, nullptr, CHANNELS, nullptr, nullptr);
    wmma_gemm<1,0,1,0><<<dim3(CHANNELS / BN, OP_DIM / BM, BATCH), 256, 0, stream>>>(
        Psibf, 0, OP_DIM,
        (const void*)dA, (long long)SPATIAL * CHANNELS, CHANNELS, SPATIAL,
        PTAbf, sPT, nullptr, nullptr, CHANNELS, nullptr, nullptr);

    // 6) corr[b] = LMask .* (PhiTMB[b] @ PhiTMA[b]^T), plus per-batch ||.||_F^2
    wmma_gemm<0,1,0,3><<<dim3(OP_DIM / BN, OP_DIM / BM, BATCH), 256, 0, stream>>>(
        PTBbf, sPT, CHANNELS, PTAbf, sPT, CHANNELS, CHANNELS,
        corrbf, sSq, nullptr, nullptr, OP_DIM, outLam, sumB);

    // 7) Z0 = corr / ||corr||_F per batch
    z0_scale<<<TAU_N / 256, 256, 0, stream>>>(corrbf, sumB, Zbf);

    // 8) batched Newton-Schulz polar of 128 x (256x256); last iter -> f32 d_out
    ushort_t* zc = Zbf;
    ushort_t* zn = Zbf2;
    const int T2 = 12;
    for (int it = 0; it < T2; ++it) {
        // G[b] = Z Z^T
        wmma_gemm<0,1,0,0><<<dim3(OP_DIM / BN, OP_DIM / BM, BATCH), 256, 0, stream>>>(
            zc, sSq, OP_DIM, zc, sSq, OP_DIM, OP_DIM,
            Gbbf, sSq, nullptr, nullptr, OP_DIM, nullptr, nullptr);
        const bool last = (it == T2 - 1);
        // Znext = 1.5Z - 0.5*(G Z)
        if (!last) {
            wmma_gemm<0,0,0,1><<<dim3(OP_DIM / BN, OP_DIM / BM, BATCH), 256, 0, stream>>>(
                Gbbf, sSq, OP_DIM, zc, sSq, OP_DIM, OP_DIM,
                zn, sSq, nullptr, zc, OP_DIM, nullptr, nullptr);
            ushort_t* t = zc; zc = zn; zn = t;
        } else {
            wmma_gemm<0,0,0,2><<<dim3(OP_DIM / BN, OP_DIM / BM, BATCH), 256, 0, stream>>>(
                Gbbf, sSq, OP_DIM, zc, sSq, OP_DIM, OP_DIM,
                nullptr, sSq, outTau, zc, OP_DIM, nullptr, nullptr);
        }
    }
}
